// NodeTransition_56023553409093
// MI455X (gfx1250) — matrix operand order repaired and checked by reading, not compiled, not run on hardware
//
#include <hip/hip_runtime.h>
#include <hip/hip_bf16.h>
#include <math.h>

// ---------------------------------------------------------------------------
// NodeTransition fused kernel for MI455X (gfx1250, wave32, WMMA bf16)
//   B=32768 tokens, C_S=H=256, C_V=128
//   d_out = [out_s (B*256) | out_v (B*128*3)] f32
// MT=32 tokens/WG: each WG reads the packed weight set exactly once (640 KB),
// B-fragments are broadcast across all M-tiles sharing a weight matrix.
// ---------------------------------------------------------------------------

typedef __bf16 bf2    __attribute__((ext_vector_type(2)));
typedef __bf16 bf8    __attribute__((ext_vector_type(8)));
typedef __bf16 v16bf  __attribute__((ext_vector_type(16)));
typedef float  v8f    __attribute__((ext_vector_type(8)));

#define NTOK   32768
#define MT     32          // tokens per workgroup
#define LDA    264         // padded LDS row stride (bf16 elems), %8==0, bank-friendly
#define NTHR   256

// packed weight offsets in d_ws (bf16 elements)
#define OFS_WS  0          // 16 nT x 8 kC x 512
#define OFS_WV  65536      // 16 x 4 x 512
#define OFS_W2  98304      // 2 x (16 x 8 x 512)
#define OFS_WOS 229376     // 16 x 8 x 512
#define OFS_WOV 294912     // 8 x 8 x 512

// ---------------------------------------------------------------------------
// Kernel 0: convert + pack f32 weights [N][K] into WMMA B-fragment-linear bf16.
// Fragment (nT,kC) = 512 bf16 as 32 lanes x 16 elems:
//   lane l, elem q  ->  W[nT*16 + (l&15)][kC*32 + (l>>4)*16 + q]
// ---------------------------------------------------------------------------
__global__ void pack_w(const float* __restrict__ src, __bf16* __restrict__ dst,
                       int K, int total, int kChunks) {
    int idx = blockIdx.x * NTHR + threadIdx.x;
    if (idx >= total) return;
    int frag = idx >> 9, e = idx & 511;
    int lane = e >> 4, q = e & 15;
    int nT = frag / kChunks, kc = frag - nT * kChunks;
    int row = nT * 16 + (lane & 15);
    int col = kc * 32 + (lane >> 4) * 16 + q;
    dst[idx] = (__bf16)src[(size_t)row * K + col];
}

// ---- fragment loaders (contiguous k = 16*(lane>>4) + elem, same order in A and B) ---------------
static __device__ inline v16bf ld_afrag(const __bf16* base, int mbase, int kb, int lane) {
    int r = lane & 15, hi = lane >> 4;
    const __bf16* p = base + (mbase + r) * LDA + kb + hi * 16;
    bf8 a0 = *(const bf8*)p;
    bf8 a1 = *(const bf8*)(p + 8);
    return __builtin_shufflevector(a0, a1, 0,1,2,3,4,5,6,7,8,9,10,11,12,13,14,15);
}
static __device__ inline v16bf ld_bfrag(const __bf16* frag, int lane) {
    const __bf16* p = frag + lane * 16;
    bf8 b0 = *(const bf8*)p;
    bf8 b1 = *(const bf8*)(p + 8);
    return __builtin_shufflevector(b0, b1, 0,1,2,3,4,5,6,7,8,9,10,11,12,13,14,15);
}

static __device__ inline v8f wmma_bf16(v16bf a, v16bf b, v8f c) {
    return __builtin_amdgcn_wmma_f32_16x16x32_bf16(false, a, false, b, (short)0, c,
                                                   false, false);
}

static __device__ inline float silu_f(float x) {
    return x * __builtin_amdgcn_rcpf(1.0f + __expf(-x));
}

// SeparableS2Activation on buf rows [m*32+tok][c]; each thread owns 2 channels
// of 16 tokens (bf16-pair LDS accesses).
static __device__ inline void sep_act(__bf16 (*buf)[LDA],
                                      const float (*TO)[4], const float (*FR)[4],
                                      int tid) {
    int half = tid >> 7;              // token half (0..1)
    int c2 = (tid & 127) * 2;         // channel pair
    for (int tk = 0; tk < 16; tk++) {
        int tok = half * 16 + tk;
        bf2 p0 = *(bf2*)&buf[tok][c2];
        bf2 p1 = *(bf2*)&buf[32 + tok][c2];
        bf2 p2 = *(bf2*)&buf[64 + tok][c2];
        bf2 p3 = *(bf2*)&buf[96 + tok][c2];
        float x0a = (float)p0[0], x0b = (float)p0[1];
        float x1a = (float)p1[0], x1b = (float)p1[1];
        float x2a = (float)p2[0], x2b = (float)p2[1];
        float x3a = (float)p3[0], x3b = (float)p3[1];
        float y1a = 0.f, y2a = 0.f, y3a = 0.f;
        float y1b = 0.f, y2b = 0.f, y3b = 0.f;
#pragma unroll
        for (int j = 0; j < 20; j++) {
            float t0 = TO[j][0], t1 = TO[j][1], t2 = TO[j][2], t3 = TO[j][3];
            float ga = t0 * x0a + t1 * x1a + t2 * x2a + t3 * x3a;
            float gb = t0 * x0b + t1 * x1b + t2 * x2b + t3 * x3b;
            float sa = silu_f(ga), sb = silu_f(gb);
            y1a = fmaf(FR[j][1], sa, y1a); y1b = fmaf(FR[j][1], sb, y1b);
            y2a = fmaf(FR[j][2], sa, y2a); y2b = fmaf(FR[j][2], sb, y2b);
            y3a = fmaf(FR[j][3], sa, y3a); y3b = fmaf(FR[j][3], sb, y3b);
        }
        bf2 o0; o0[0] = (__bf16)silu_f(x0a); o0[1] = (__bf16)silu_f(x0b);
        bf2 o1; o1[0] = (__bf16)y1a;         o1[1] = (__bf16)y1b;
        bf2 o2; o2[0] = (__bf16)y2a;         o2[1] = (__bf16)y2b;
        bf2 o3; o3[0] = (__bf16)y3a;         o3[1] = (__bf16)y3b;
        *(bf2*)&buf[tok][c2]      = o0;
        *(bf2*)&buf[32 + tok][c2] = o1;
        *(bf2*)&buf[64 + tok][c2] = o2;
        *(bf2*)&buf[96 + tok][c2] = o3;
    }
}

static __device__ inline float red16(float x) {
#pragma unroll
    for (int m = 8; m >= 1; m >>= 1) x += __shfl_xor(x, m, 16);
    return x;
}

// ---------------------------------------------------------------------------
__global__ __launch_bounds__(NTHR)
void node_transition_fused(const float* __restrict__ s, const float* __restrict__ v,
                           const float* __restrict__ bs, const float* __restrict__ b2,
                           const float* __restrict__ bos,
                           const float* __restrict__ gam_s, const float* __restrict__ bet_s,
                           const float* __restrict__ gam_v,
                           const __bf16* __restrict__ Wp, float* __restrict__ out) {
    // act buffer X (staging / GEMM2-out) : 128 x 264 bf16 = 67.6 KB
    __shared__ __bf16 smX[128][LDA];
    // union region: act buffer Y (GEMM1-out) 67.6 KB  OR  f32 out staging 80 KB
    __shared__ __align__(16) char smU[(32 * 256 + 96 * 128) * 4];
    __bf16 (*smY)[LDA] = reinterpret_cast<__bf16 (*)[LDA]>(smU);
    float* smOS = reinterpret_cast<float*>(smU);      // [32][256]
    float* smOV = smOS + 32 * 256;                    // [96][128]
    __shared__ float sTO[20][4], sFR[20][4];

    const int tid  = threadIdx.x;
    const int wave = __builtin_amdgcn_readfirstlane(tid >> 5);   // uniform wave id
    const int lane = tid & 31;
    const int t0   = blockIdx.x * MT;

    // --- spherical-grid quadrature tables (l<=1, RES_B=4, RES_A=5) ---------
    if (tid < 20) {
        const float xb[4]  = {-0.8611363115940526f, -0.3399810435848563f,
                               0.3399810435848563f,  0.8611363115940526f};
        const float wbq[4] = { 0.3478548451374538f,  0.6521451548625461f,
                               0.6521451548625461f,  0.3478548451374538f};
        int b = tid / 5, a = tid - b * 5;
        float sb = sqrtf(1.0f - xb[b] * xb[b]);
        float al = 6.283185307179586f * (float)a / 5.0f;
        const float wa = 6.283185307179586f / 5.0f;
        const float k0 = 0.28209479177387814f, k1 = 0.48860251190291992f;
        float t[4] = {k0, k1 * sb * sinf(al), k1 * xb[b], k1 * sb * cosf(al)};
        float w = wbq[b] * wa;
#pragma unroll
        for (int i = 0; i < 4; i++) { sTO[tid][i] = t[i]; sFR[tid][i] = t[i] * w; }
    }

    // --- stage inputs as bf16 A-matrix: rows [m*32+tok][k] -----------------
    for (int i = 0; i < 32; i++) {               // s -> rows 0..31
        int idx = i * NTHR + tid, tok = idx >> 8, k = idx & 255;
        smX[tok][k] = (__bf16)s[(size_t)(t0 + tok) * 256 + k];
    }
    for (int i = 0; i < 48; i++) {               // v^T -> rows 32..127
        int idx = i * NTHR + tid, c = idx & 127, rv = idx >> 7;
        int mv = rv >> 5, tok = rv & 31;
        smX[32 + rv][c] = (__bf16)v[(size_t)(t0 + tok) * 384 + c * 3 + mv];
    }
    __syncthreads();

    // ---- GEMM1a: embed_s = s @ Ws^T + bs (m=0, 2 token-tiles per B-load) --
    for (int nT = wave; nT < 16; nT += 8) {
        const __bf16* frag = Wp + OFS_WS + (size_t)(nT * 8) * 512;
        v8f a0 = {}, a1 = {};
#pragma unroll
        for (int kc = 0; kc < 8; kc++) {
            v16bf b = ld_bfrag(frag + kc * 512, lane);
            a0 = wmma_bf16(ld_afrag(&smX[0][0],  0, kc * 32, lane), b, a0);
            a1 = wmma_bf16(ld_afrag(&smX[0][0], 16, kc * 32, lane), b, a1);
        }
        int n = nT * 16 + (lane & 15), mh = (lane >> 4) * 8;
        float bv = bs[n];
#pragma unroll
        for (int g = 0; g < 8; g++) {
            smY[mh + g][n]      = (__bf16)(a0[g] + bv);
            smY[16 + mh + g][n] = (__bf16)(a1[g] + bv);
        }
    }
    // ---- GEMM1b: embed_v = v^T @ Wv^T (m=1..3, 6 tiles share each B-load) -
    for (int nT = wave; nT < 16; nT += 8) {
        const __bf16* frag = Wp + OFS_WV + (size_t)(nT * 4) * 512;
        v8f acc[6] = {};
#pragma unroll
        for (int kc = 0; kc < 4; kc++) {
            v16bf b = ld_bfrag(frag + kc * 512, lane);
#pragma unroll
            for (int q = 0; q < 6; q++)
                acc[q] = wmma_bf16(ld_afrag(&smX[0][0], 32 + q * 16, kc * 32, lane),
                                   b, acc[q]);
        }
        int n = nT * 16 + (lane & 15), mh = (lane >> 4) * 8;
#pragma unroll
        for (int q = 0; q < 6; q++)
#pragma unroll
            for (int g = 0; g < 8; g++)
                smY[32 + q * 16 + mh + g][n] = (__bf16)acc[q][g];
    }
    __syncthreads();
    sep_act(smY, sTO, sFR, tid);
    __syncthreads();

    // ---- GEMM2: W2[0] for m=0, W2[1] for m=1..3 (8 tiles per 2 B-loads) ---
    for (int nT = wave; nT < 16; nT += 8) {
        const __bf16* f0 = Wp + OFS_W2 + (size_t)(nT * 8) * 512;
        const __bf16* f1 = f0 + 65536;
        v8f acc[8] = {};
#pragma unroll
        for (int kc = 0; kc < 8; kc++) {
            v16bf b0 = ld_bfrag(f0 + kc * 512, lane);
            v16bf b1 = ld_bfrag(f1 + kc * 512, lane);
#pragma unroll
            for (int q = 0; q < 8; q++)   // q = m*2 + tt -> row base q*16
                acc[q] = wmma_bf16(ld_afrag(&smY[0][0], q * 16, kc * 32, lane),
                                   (q < 2) ? b0 : b1, acc[q]);
        }
        int n = nT * 16 + (lane & 15), mh = (lane >> 4) * 8;
        float bv = b2[n];
#pragma unroll
        for (int q = 0; q < 8; q++)
#pragma unroll
            for (int g = 0; g < 8; g++)
                smX[q * 16 + mh + g][n] = (__bf16)(acc[q][g] + (q < 2 ? bv : 0.0f));
    }
    __syncthreads();
    sep_act(smX, sTO, sFR, tid);
    __syncthreads();

    // ---- GEMM3a: out_s = emb0 @ Wos^T + bos -------------------------------
    for (int nT = wave; nT < 16; nT += 8) {
        const __bf16* frag = Wp + OFS_WOS + (size_t)(nT * 8) * 512;
        v8f a0 = {}, a1 = {};
#pragma unroll
        for (int kc = 0; kc < 8; kc++) {
            v16bf b = ld_bfrag(frag + kc * 512, lane);
            a0 = wmma_bf16(ld_afrag(&smX[0][0],  0, kc * 32, lane), b, a0);
            a1 = wmma_bf16(ld_afrag(&smX[0][0], 16, kc * 32, lane), b, a1);
        }
        int n = nT * 16 + (lane & 15), mh = (lane >> 4) * 8;
        float bv = bos[n];
#pragma unroll
        for (int g = 0; g < 8; g++) {
            smOS[(mh + g) * 256 + n]      = a0[g] + bv;
            smOS[(16 + mh + g) * 256 + n] = a1[g] + bv;
        }
    }
    // ---- GEMM3b: out_v = emb(1..3) @ Wov^T (one nT per wave) --------------
    {
        int nT = wave;   // 8 waves, 8 N-tiles of 128 channels
        const __bf16* frag = Wp + OFS_WOV + (size_t)(nT * 8) * 512;
        v8f acc[6] = {};
#pragma unroll
        for (int kc = 0; kc < 8; kc++) {
            v16bf b = ld_bfrag(frag + kc * 512, lane);
#pragma unroll
            for (int q = 0; q < 6; q++)
                acc[q] = wmma_bf16(ld_afrag(&smX[0][0], 32 + q * 16, kc * 32, lane),
                                   b, acc[q]);
        }
        int n = nT * 16 + (lane & 15), mh = (lane >> 4) * 8;
#pragma unroll
        for (int q = 0; q < 6; q++)
#pragma unroll
            for (int g = 0; g < 8; g++)
                smOV[(q * 16 + mh + g) * 128 + n] = acc[q][g];
    }
    __syncthreads();

    // ---- epilogue: LayerNorm(s + out_s) + equivariant vector norm ---------
    for (int pass = 0; pass < 2; pass++) {
        int tok = pass * 16 + (tid >> 4), i = tid & 15;
        size_t gt = (size_t)(t0 + tok);

        float xs[16], sum = 0.f, sum2 = 0.f;
#pragma unroll
        for (int j = 0; j < 16; j++) {
            int c = i + 16 * j;
            float x = s[gt * 256 + c] + smOS[tok * 256 + c];
            xs[j] = x; sum += x; sum2 += x * x;
        }
        sum = red16(sum); sum2 = red16(sum2);
        float mu = sum * (1.0f / 256.0f);
        float var = sum2 * (1.0f / 256.0f) - mu * mu;
        float rs = rsqrtf(var + 1e-5f);
#pragma unroll
        for (int j = 0; j < 16; j++) {
            int c = i + 16 * j;
            out[gt * 256 + c] = (xs[j] - mu) * rs * gam_s[c] + bet_s[c];
        }

        float vr[8][3], ss = 0.f;
#pragma unroll
        for (int j = 0; j < 8; j++) {
            int c = i + 16 * j;
#pragma unroll
            for (int mv = 0; mv < 3; mv++) {
                float x = v[gt * 384 + c * 3 + mv] + smOV[(mv * 32 + tok) * 128 + c];
                vr[j][mv] = x; ss += x * x;
            }
        }
        ss = red16(ss);
        float inv = rsqrtf(ss * (1.0f / 128.0f) + 1e-5f);
        float* outv = out + (size_t)NTOK * 256;
#pragma unroll
        for (int j = 0; j < 8; j++) {
            int c = i + 16 * j;
            float g = gam_v[c] * inv;
#pragma unroll
            for (int mv = 0; mv < 3; mv++)
                outv[gt * 384 + c * 3 + mv] = vr[j][mv] * g;
        }
    }
}

// ---------------------------------------------------------------------------
extern "C" void kernel_launch(void* const* d_in, const int* in_sizes, int n_in,
                              void* d_out, int out_size, void* d_ws, size_t ws_size,
                              hipStream_t stream) {
    (void)in_sizes; (void)n_in; (void)out_size; (void)ws_size;
    const float* s   = (const float*)d_in[0];
    const float* v   = (const float*)d_in[1];
    const float* Ws  = (const float*)d_in[2];
    const float* bs  = (const float*)d_in[3];
    const float* Wv  = (const float*)d_in[4];
    const float* W2  = (const float*)d_in[5];
    const float* b2  = (const float*)d_in[6];
    const float* Wos = (const float*)d_in[7];
    const float* bos = (const float*)d_in[8];
    const float* Wov = (const float*)d_in[9];
    const float* gs  = (const float*)d_in[10];
    const float* bt  = (const float*)d_in[11];
    const float* gv  = (const float*)d_in[12];
    __bf16* wp = (__bf16*)d_ws;

    pack_w<<<256, NTHR, 0, stream>>>(Ws,  wp + OFS_WS,  256, 65536,  8);
    pack_w<<<128, NTHR, 0, stream>>>(Wv,  wp + OFS_WV,  128, 32768,  4);
    pack_w<<<512, NTHR, 0, stream>>>(W2,  wp + OFS_W2,  256, 131072, 8);
    pack_w<<<256, NTHR, 0, stream>>>(Wos, wp + OFS_WOS, 256, 65536,  8);
    pack_w<<<128, NTHR, 0, stream>>>(Wov, wp + OFS_WOV, 256, 32768,  8);

    node_transition_fused<<<NTOK / MT, NTHR, 0, stream>>>(
        s, v, bs, b2, bos, gs, bt, gv, wp, (float*)d_out);
}
